// bhattacharya_rnn_18476949307426
// MI455X (gfx1250) — compile-verified, matrix operand order checked by reading, not run on hardware
//
#include <hip/hip_runtime.h>
#include <hip/hip_bf16.h>
#include <hip/hip_fp16.h>

// MI455X / gfx1250, wave32. All GEMMs via v_wmma_f32_16x16x32_f16.
// Mixed precision: f16 operands, f32 accumulation, f32 recurrent state.

typedef __attribute__((ext_vector_type(16))) _Float16 v16h;
typedef __attribute__((ext_vector_type(8)))  _Float16 v8h;
typedef __attribute__((ext_vector_type(8)))  float    v8f;
typedef __attribute__((ext_vector_type(4)))  unsigned int u32x4;
typedef __attribute__((ext_vector_type(8)))  int      i32x8;
typedef __attribute__((ext_vector_type(4)))  int      i32x4;

#define NSV   20
#define BATCH 256
#define SEQ   1000

__device__ __forceinline__ float selu_f(float x) {
    const float sc = 1.0507009873554805f;
    const float al = 1.6732632423543772f;
    return sc * (x > 0.f ? x : al * (__expf(x) - 1.f));
}

__device__ __forceinline__ v8f wmma16(v16h a, v16h b, v8f c) {
    // D = A(16x32 f16) * B(32x16 f16) + C(16x16 f32)
    return __builtin_amdgcn_wmma_f32_16x16x32_f16(
        /*neg_a=*/false, a, /*neg_b=*/false, b,
        /*c_mod=*/(short)0, c, /*reuse_a=*/false, /*reuse_b=*/false);
}

// A-fragment: 16x32 tile from row-major f16 buffer (row stride `ld` halves).
// ISA 16-bit A layout: lane g=0: halves 0-7 -> K 0..7, 8-15 -> K 16..23;
//                      lane g=1: halves 0-7 -> K 8..15, 8-15 -> K 24..31
__device__ __forceinline__ v16h load_fragA(const _Float16* H, int ld, int k0, int lane) {
    int m = lane & 15, g = lane >> 4;
    const _Float16* p = H + (size_t)m * ld + k0 + g * 8;
    v8h lo = *(const v8h*)p;
    v8h hi = *(const v8h*)(p + 16);
    return __builtin_shufflevector(lo, hi, 0,1,2,3,4,5,6,7,8,9,10,11,12,13,14,15);
}

// B-fragment: 32x16 tile. Weights stored row-major W[n][k]; lanes index n,
// K mapping mirrors the A layout.
__device__ __forceinline__ v16h load_fragB(const _Float16* W, int ld, int n0, int k0, int lane) {
    int n = n0 + (lane & 15), g = lane >> 4;
    const _Float16* p = W + (size_t)n * ld + k0 + g * 8;
    v8h lo = *(const v8h*)p;
    v8h hi = *(const v8h*)(p + 16);
    return __builtin_shufflevector(lo, hi, 0,1,2,3,4,5,6,7,8,9,10,11,12,13,14,15);
}

// ---------------------------------------------------------------- pack weights
__global__ void pack_w_f16(const float* __restrict__ src, _Float16* __restrict__ dst,
                           int dout, int din, int dout_pad, int din_pad) {
    int idx = blockIdx.x * blockDim.x + threadIdx.x;
    int total = dout_pad * din_pad;
    if (idx >= total) return;
    int n = idx / din_pad, k = idx - n * din_pad;
    float v = (n < dout && k < din) ? src[(size_t)n * din + k] : 0.f;
    dst[idx] = (_Float16)v;
}

// ----------------------------------------------------- phase 1: recurrent scan
// One 16-row batch tile per block; 8 waves, one 16-wide n-tile per wave.
__device__ __forceinline__ void layer128(const _Float16* hsrc, _Float16* hdst,
                                         const _Float16* W, const float* bias128,
                                         int nt, int lane) {
    v16h a0 = load_fragA(hsrc, 128,  0, lane);
    v16h a1 = load_fragA(hsrc, 128, 32, lane);
    v16h a2 = load_fragA(hsrc, 128, 64, lane);
    v16h a3 = load_fragA(hsrc, 128, 96, lane);
    v8f c = {};
    c = wmma16(a0, load_fragB(W, 128, nt * 16,  0, lane), c);
    c = wmma16(a1, load_fragB(W, 128, nt * 16, 32, lane), c);
    c = wmma16(a2, load_fragB(W, 128, nt * 16, 64, lane), c);
    c = wmma16(a3, load_fragB(W, 128, nt * 16, 96, lane), c);
    int n = nt * 16 + (lane & 15);
    float bias = bias128[n];
    int g = lane >> 4;
    #pragma unroll
    for (int v = 0; v < 8; ++v)
        hdst[(v + 8 * g) * 128 + n] = (_Float16)selu_f(c[v] + bias);
}

__global__ __launch_bounds__(256)
void rnn_scan_kernel(const float* __restrict__ x,
                     const _Float16* __restrict__ gwall,   // GW0|GW1|GW2|GW3 contiguous f16
                     const float* __restrict__ gb0, const float* __restrict__ gb1,
                     const float* __restrict__ gb2, const float* __restrict__ gb3,
                     _Float16* __restrict__ finput) {
    extern __shared__ char smem[];
    _Float16* gw0L = (_Float16*)smem;            // 128*32   @ LDS byte 0
    _Float16* gw1L = gw0L + 128 * 32;            // 128*128
    _Float16* gw2L = gw1L + 128 * 128;           // 128*128
    _Float16* gw3L = gw2L + 128 * 128;           // 32*128   (weights end @ 81920 B)
    _Float16* hin  = gw3L + 32 * 128;            // 16*32
    _Float16* hA   = hin  + 16 * 32;             // 16*128
    _Float16* hB   = hA   + 16 * 128;            // 16*128
    float*    svf  = (float*)(hB + 16 * 128);    // 16*NSV (f32 recurrent state)
    float*    dtl  = svf + 16 * NSV;             // 16
    float*    bb0  = dtl + 16;                   // 128
    float*    bb1  = bb0 + 128;                  // 128
    float*    bb2  = bb1 + 128;                  // 128
    float*    bb3  = bb2 + 128;                  // 32

    const int tid  = threadIdx.x;
    const int lane = tid & 31;
    const int wave = tid >> 5;
    const int r0   = blockIdx.x * 16;

    // --- TDM: stage the 80KB g-weight block into LDS as a 2D tile (256x40 of
    // 8B units, stride 256). D# per CDNA5 ISA §8: group0 = {flags, lds_addr,
    // global_addr, type=2}; group1 = dims/strides. Followed by a cooperative
    // copy of the same bytes after s_wait_tensorcnt, which pins the final LDS
    // contents deterministically (TDM semantics are probe-verified for compile
    // only).
    if (tid == 0) {
        unsigned long long ga = (unsigned long long)(const void*)gwall;
        u32x4 g0;
        g0[0] = 1u;                                              // count=1, user desc
        g0[1] = 0u;                                              // lds_addr = 0
        g0[2] = (unsigned)(ga & 0xFFFFFFFFu);                    // global_addr[31:0]
        g0[3] = (unsigned)((ga >> 32) & 0x1FFFFFFu) | (2u << 30); // addr[56:32] | type=2
        i32x8 g1;
        g1[0] = (3 << 16);            // workgroup_mask=0, data_size=8B
        g1[1] = (int)(256u << 16);    // tensor_dim0 = 256 units (low16 in [63:48])
        g1[2] = (int)(40u << 16);     // tensor_dim0 hi=0 | tensor_dim1 = 40
        g1[3] = (int)(256u << 16);    // tensor_dim1 hi=0 | tile_dim0 = 256
        g1[4] = 40;                   // tile_dim1 = 40 | tile_dim2 = 0
        g1[5] = 256;                  // tensor_dim0_stride = 256 (low 32)
        g1[6] = 0;                    // stride0 hi | stride1 lo
        g1[7] = 0;                    // stride1 hi
        i32x4 gz = {0, 0, 0, 0};
#if defined(__clang_major__) && (__clang_major__ >= 23)
        i32x8 gz8 = {0, 0, 0, 0, 0, 0, 0, 0};
        __builtin_amdgcn_tensor_load_to_lds(g0, g1, gz, gz, gz8, 0);
#else
        __builtin_amdgcn_tensor_load_to_lds(g0, g1, gz, gz, 0);
#endif
        __builtin_amdgcn_s_wait_tensorcnt(0);
    }
    __syncthreads();

    // deterministic weight + bias staging, zero the recurrent state
    {
        const uint4* src = (const uint4*)gwall;
        uint4* dst = (uint4*)smem;
        const int nchunk = (40960 * 2) / 16;     // 5120 x 16B
        for (int i = tid; i < nchunk; i += 256) dst[i] = src[i];
        for (int i = tid; i < 16 * NSV; i += 256) svf[i] = 0.f;
        if (tid < 128) { bb0[tid] = gb0[tid]; bb1[tid] = gb1[tid]; bb2[tid] = gb2[tid]; }
        if (tid < NSV) bb3[tid] = gb3[tid];
    }
    __syncthreads();

    for (int t = 0; t < SEQ - 1; ++t) {
        // (a) lanes 0-15 build the 16x32 WMMA input tile; lanes 16-31 emit the
        //     f-net input row for step t (strain | strain_dot | sv) in parallel.
        if (tid < 16) {
            int m = tid, r = r0 + m;
            const float* xr = x + ((size_t)r * SEQ + t) * 13;
            dtl[m] = xr[13] - xr[0];             // time[t+1] - time[t]
            _Float16* hm = hin + m * 32;
            #pragma unroll
            for (int j = 0; j < 6; ++j) hm[j] = (_Float16)xr[1 + j];
            #pragma unroll
            for (int j = 0; j < NSV; ++j) hm[6 + j] = (_Float16)svf[m * NSV + j];
            #pragma unroll
            for (int j = 26; j < 32; ++j) hm[j] = (_Float16)0.f;
        } else if (tid < 32) {
            int m = tid - 16, r = r0 + m;
            const float* xr = x + ((size_t)r * SEQ + t) * 13;
            _Float16* fi = finput + ((size_t)r * SEQ + t) * 32;
            #pragma unroll
            for (int j = 0; j < 12; ++j) fi[j] = (_Float16)xr[1 + j];
            #pragma unroll
            for (int j = 0; j < NSV; ++j) fi[12 + j] = (_Float16)svf[m * NSV + j];
        }
        __syncthreads();

        // (b) layer0: 26(->32) -> 128, one n-tile per wave
        {
            v16h a = load_fragA(hin, 32, 0, lane);
            v16h b = load_fragB(gw0L, 32, wave * 16, 0, lane);
            v8f c = {};
            c = wmma16(a, b, c);
            int n = wave * 16 + (lane & 15);
            float bias = bb0[n];
            int g = lane >> 4;
            #pragma unroll
            for (int v = 0; v < 8; ++v)
                hA[(v + 8 * g) * 128 + n] = (_Float16)selu_f(c[v] + bias);
        }
        __syncthreads();

        layer128(hA, hB, gw1L, bb1, wave, lane);   // (c) 128 -> 128
        __syncthreads();
        layer128(hB, hA, gw2L, bb2, wave, lane);   // (d) 128 -> 128
        __syncthreads();

        // (e) layer3: 128 -> 20(->32), then sv += dt * svd  (waves 0,1)
        if (wave < 2) {
            v16h a0 = load_fragA(hA, 128,  0, lane);
            v16h a1 = load_fragA(hA, 128, 32, lane);
            v16h a2 = load_fragA(hA, 128, 64, lane);
            v16h a3 = load_fragA(hA, 128, 96, lane);
            v8f c = {};
            c = wmma16(a0, load_fragB(gw3L, 128, wave * 16,  0, lane), c);
            c = wmma16(a1, load_fragB(gw3L, 128, wave * 16, 32, lane), c);
            c = wmma16(a2, load_fragB(gw3L, 128, wave * 16, 64, lane), c);
            c = wmma16(a3, load_fragB(gw3L, 128, wave * 16, 96, lane), c);
            int n = wave * 16 + (lane & 15);
            if (n < NSV) {
                float bias = bb3[n];
                int g = lane >> 4;
                #pragma unroll
                for (int v = 0; v < 8; ++v) {
                    int m = v + 8 * g;
                    svf[m * NSV + n] += dtl[m] * (c[v] + bias);
                }
            }
        }
        __syncthreads();
    }

    // epilogue: f-net input row S-1 uses the final state
    if (tid < 16) {
        int m = tid, r = r0 + m;
        const float* xr = x + ((size_t)r * SEQ + (SEQ - 1)) * 13;
        _Float16* fi = finput + ((size_t)r * SEQ + (SEQ - 1)) * 32;
        #pragma unroll
        for (int j = 0; j < 12; ++j) fi[j] = (_Float16)xr[1 + j];
        #pragma unroll
        for (int j = 0; j < NSV; ++j) fi[12 + j] = (_Float16)svf[m * NSV + j];
    }
}

// --------------------------------------------------------- phase 2: f-net GEMM
// 32 rows (2 m-tiles) per wave: every B-fragment feeds two WMMAs, halving the
// per-row L2 weight traffic.
__device__ __forceinline__ void mlp256x2(const _Float16* hsrc, _Float16* hdst,
                                         const _Float16* __restrict__ W,
                                         const float* __restrict__ bvec, int lane) {
    v16h a0[8], a1[8];
    #pragma unroll
    for (int kt = 0; kt < 8; ++kt) {
        a0[kt] = load_fragA(hsrc,            256, kt * 32, lane);
        a1[kt] = load_fragA(hsrc + 16 * 256, 256, kt * 32, lane);
    }
    for (int nt = 0; nt < 16; ++nt) {
        v8f c0 = {}, c1 = {};
        #pragma unroll
        for (int kt = 0; kt < 8; ++kt) {
            v16h b = load_fragB(W, 256, nt * 16, kt * 32, lane);
            c0 = wmma16(a0[kt], b, c0);
            c1 = wmma16(a1[kt], b, c1);
        }
        int n = nt * 16 + (lane & 15);
        float bias = bvec[n];
        int g = lane >> 4;
        #pragma unroll
        for (int v = 0; v < 8; ++v) {
            hdst[(v + 8 * g) * 256 + n]        = (_Float16)selu_f(c0[v] + bias);
            hdst[(16 + v + 8 * g) * 256 + n]   = (_Float16)selu_f(c1[v] + bias);
        }
    }
}

__global__ __launch_bounds__(32)
void mlp_head_kernel(const _Float16* __restrict__ finput,
                     const _Float16* __restrict__ fw0h, const float* __restrict__ fb0,
                     const _Float16* __restrict__ fw1h, const float* __restrict__ fb1,
                     const _Float16* __restrict__ fw2h, const float* __restrict__ fb2,
                     const _Float16* __restrict__ fw3h, const float* __restrict__ fb3,
                     float* __restrict__ out) {
    extern __shared__ char smem[];
    _Float16* h1 = (_Float16*)smem;      // 32*256
    _Float16* h2 = h1 + 32 * 256;        // 32*256

    const int lane = threadIdx.x & 31;
    const long row0 = (long)blockIdx.x * 32;

    // hint the L2-resident weight streams (global_prefetch_b8)
    __builtin_prefetch(fw1h + (size_t)lane * 2048, 0, 1);
    __builtin_prefetch(fw2h + (size_t)lane * 2048, 0, 1);

    // layer0: K=32 exactly; A-fragments stream straight from global finput
    {
        v16h a0 = load_fragA(finput + row0 * 32,        32, 0, lane);
        v16h a1 = load_fragA(finput + (row0 + 16) * 32, 32, 0, lane);
        for (int nt = 0; nt < 16; ++nt) {
            v16h b = load_fragB(fw0h, 32, nt * 16, 0, lane);
            v8f c0 = {}, c1 = {};
            c0 = wmma16(a0, b, c0);
            c1 = wmma16(a1, b, c1);
            int n = nt * 16 + (lane & 15);
            float bias = fb0[n];
            int g = lane >> 4;
            #pragma unroll
            for (int v = 0; v < 8; ++v) {
                h1[(v + 8 * g) * 256 + n]      = (_Float16)selu_f(c0[v] + bias);
                h1[(16 + v + 8 * g) * 256 + n] = (_Float16)selu_f(c1[v] + bias);
            }
        }
    }
    __syncthreads();
    mlp256x2(h1, h2, fw1h, fb1, lane);
    __syncthreads();
    mlp256x2(h2, h1, fw2h, fb2, lane);
    __syncthreads();

    // layer3: 256 -> 6 (padded to one 16-wide tile), both m-tiles
    {
        v8f c0 = {}, c1 = {};
        #pragma unroll
        for (int kt = 0; kt < 8; ++kt) {
            v16h b = load_fragB(fw3h, 256, 0, kt * 32, lane);
            c0 = wmma16(load_fragA(h1,            256, kt * 32, lane), b, c0);
            c1 = wmma16(load_fragA(h1 + 16 * 256, 256, kt * 32, lane), b, c1);
        }
        int n = lane & 15, g = lane >> 4;
        if (n < 6) {
            float bias = fb3[n];
            #pragma unroll
            for (int v = 0; v < 8; ++v) {
                long m0 = row0 + v + 8 * g;
                out[m0 * 6 + n]        = c0[v] + bias;
                out[(m0 + 16) * 6 + n] = c1[v] + bias;
            }
        }
    }
}

// ---------------------------------------------------------------------- launch
extern "C" void kernel_launch(void* const* d_in, const int* in_sizes, int n_in,
                              void* d_out, int out_size, void* d_ws, size_t ws_size,
                              hipStream_t stream) {
    const float* x   = (const float*)d_in[0];
    const float* gw0 = (const float*)d_in[1];  const float* gb0 = (const float*)d_in[2];
    const float* gw1 = (const float*)d_in[3];  const float* gb1 = (const float*)d_in[4];
    const float* gw2 = (const float*)d_in[5];  const float* gb2 = (const float*)d_in[6];
    const float* gw3 = (const float*)d_in[7];  const float* gb3 = (const float*)d_in[8];
    const float* fw0 = (const float*)d_in[9];  const float* fb0 = (const float*)d_in[10];
    const float* fw1 = (const float*)d_in[11]; const float* fb1 = (const float*)d_in[12];
    const float* fw2 = (const float*)d_in[13]; const float* fb2 = (const float*)d_in[14];
    const float* fw3 = (const float*)d_in[15]; const float* fb3 = (const float*)d_in[16];

    // workspace layout (needs ~16.9 MB):
    //   [0, 368640)          : packed f16 weights (padded, row-major)
    //   [512KB, 512KB+16MB)  : f-net input rows, f16 [B*S][32]
    char* ws = (char*)d_ws;
    _Float16* W = (_Float16*)ws;
    const size_t GW0 = 0,     GW1 = 4096,  GW2 = 20480,  GW3 = 36864;
    const size_t FW0 = 40960, FW1 = 49152, FW2 = 114688, FW3 = 180224;
    _Float16* finput = (_Float16*)(ws + (512u << 10));

    auto pack = [&](const float* src, size_t off, int dout, int din, int dp, int kp) {
        int total = dp * kp;
        pack_w_f16<<<(total + 255) / 256, 256, 0, stream>>>(src, W + off, dout, din, dp, kp);
    };
    pack(gw0, GW0, 128,  26, 128,  32);
    pack(gw1, GW1, 128, 128, 128, 128);
    pack(gw2, GW2, 128, 128, 128, 128);
    pack(gw3, GW3,  20, 128,  32, 128);
    pack(fw0, FW0, 256,  32, 256,  32);
    pack(fw1, FW1, 256, 256, 256, 256);
    pack(fw2, FW2, 256, 256, 256, 256);
    pack(fw3, FW3,   6, 256,  16, 256);

    // phase 1: 16 batch tiles x 8 waves; g-weights + biases resident in LDS
    size_t lds1 = (size_t)(128*32 + 128*128 + 128*128 + 32*128 + 16*32 + 2*16*128) * sizeof(_Float16)
                + (size_t)(16 * NSV + 16 + 3 * 128 + 32) * sizeof(float);
    rnn_scan_kernel<<<16, 256, lds1, stream>>>(x, W + GW0, gb0, gb1, gb2, gb3, finput);

    // phase 2: 256000 rows / 32 per block (2 m-tiles per wave)
    size_t lds2 = (size_t)(2 * 32 * 256) * sizeof(_Float16);
    mlp_head_kernel<<<(BATCH * SEQ) / 32, 32, lds2, stream>>>(
        finput, W + FW0, fb0, W + FW1, fb1, W + FW2, fb2, W + FW3, fb3, (float*)d_out);
}
